// FFTLinearLayerV2_51857435132042
// MI455X (gfx1250) — matrix operand order repaired and checked by reading, not compiled, not run on hardware
//
#include <hip/hip_runtime.h>

// FFT-as-GEMM for MI455X (gfx1250, wave32, WMMA f16->f32).
// Y[o,m] = sum_k W[o,k] * X[k,m];  o,k in [0,512), m = (b,h,w) in [0, 262144)

typedef __attribute__((ext_vector_type(16))) _Float16     v16h;
typedef __attribute__((ext_vector_type(2)))  __fp16       v2fp16;
typedef __attribute__((ext_vector_type(8)))  float        v8f;
typedef __attribute__((ext_vector_type(4)))  float        v4f;
typedef __attribute__((ext_vector_type(4)))  unsigned int v4u;

#define TM    128        // o tile per workgroup
#define TN    128        // m tile per workgroup
#define TK    32         // k step (WMMA f16 K)
#define KTOT  512        // 2N
#define NCH   256        // N (channels)
#define NB    4          // batch
#define HW    65536      // H*W
#define PITCH 40         // halves per LDS row (32 data + 8 pad) = 80B, 16B aligned

union Frag16 { v16h h; v4u u[2]; };
union Acc    { v8f  v; float f[8]; };

// One-instruction convert+pack: v_cvt_pkrtz_f16_f32
static __device__ inline unsigned int pkh2(float a, float b) {
    v2fp16 h = __builtin_amdgcn_cvt_pkrtz(a, b);
    return __builtin_bit_cast(unsigned int, h);
}

__global__ __launch_bounds__(256)
void fft_dft_gemm_f16wmma(const float* __restrict__ xr,
                          const float* __restrict__ xi,
                          const float* __restrict__ w,
                          float* __restrict__ out)
{
    __shared__ _Float16 sA[2][TM * PITCH];   // W tile  [o_local][k_local]
    __shared__ _Float16 sB[2][TN * PITCH];   // X tile  [m_local][k_local] (transposed)

    const int tid  = threadIdx.x;
    const int lane = tid & 31;
    const int wave = tid >> 5;
    const int wo   = wave & 3;     // 4 waves along o, 32 rows each
    const int wm   = wave >> 2;    // 2 waves along m, 64 cols each
    const int lhi  = lane >> 4;    // 0/1 half of wave
    const int llo  = lane & 15;

    const int obase = blockIdx.x * TM;       // 0..384 step 128 (never straddles o=256)
    const int mbase = blockIdx.y * TN;
    const int bidx  = mbase >> 16;           // HW = 65536 -> tile lies in one batch
    const int rbase = mbase & (HW - 1);

    // staging decomposition
    const int wq  = tid & 7;                 // float4 slot within a W row (8*4 = 32 k)
    const int wr  = tid >> 3;                // W row 0..31 (x4 passes -> 128)
    const int xm4 = tid & 31;                // float4 slot along m (32*4 = 128 m)
    const int xk  = tid >> 5;                // k-pair 0..7 (x2 passes -> 16 pairs = 32 k)

    Acc acc[2][4];
#pragma unroll
    for (int f = 0; f < 2; ++f)
#pragma unroll
        for (int mf = 0; mf < 4; ++mf)
#pragma unroll
            for (int v = 0; v < 8; ++v) acc[f][mf].f[v] = 0.0f;

    v4f wreg[4];         // next-stage W data
    v4f xreg[2][2];      // next-stage X data: [pass][k / k+1]

    const float* xrb = xr + (size_t)bidx * NCH * HW + rbase;
    const float* xib = xi + (size_t)bidx * NCH * HW + rbase;

    auto loadStage = [&](int kb) {
        const float* ws = w + (size_t)obase * KTOT + kb;
#pragma unroll
        for (int p = 0; p < 4; ++p)
            wreg[p] = *(const v4f*)(ws + (size_t)(wr + 32 * p) * KTOT + wq * 4);
        const float* xsrc = (kb < NCH) ? xrb : xib;   // tile never straddles 256
        const int kk = kb & (NCH - 1);
#pragma unroll
        for (int p = 0; p < 2; ++p) {
            const int k0 = 2 * (xk + 8 * p);
            xreg[p][0] = *(const v4f*)(xsrc + (size_t)(kk + k0) * HW + xm4 * 4);
            xreg[p][1] = *(const v4f*)(xsrc + (size_t)(kk + k0 + 1) * HW + xm4 * 4);
        }
    };

    auto storeStage = [&](int buf) {
        _Float16* A = &sA[buf][0];
#pragma unroll
        for (int p = 0; p < 4; ++p) {
            unsigned long long pk =
                (unsigned long long)pkh2(wreg[p].x, wreg[p].y) |
                ((unsigned long long)pkh2(wreg[p].z, wreg[p].w) << 32);
            *(unsigned long long*)(A + (wr + 32 * p) * PITCH + wq * 4) = pk;
        }
        _Float16* Bt = &sB[buf][0];
#pragma unroll
        for (int p = 0; p < 2; ++p) {
            const int k0 = 2 * (xk + 8 * p);
#pragma unroll
            for (int j = 0; j < 4; ++j) {
                // pack (k, k+1) for one m into one dword
                *(unsigned int*)(Bt + (xm4 * 4 + j) * PITCH + k0) =
                    pkh2(xreg[p][0][j], xreg[p][1][j]);
            }
        }
    };

    // A frag (16x32 f16): lane holds row llo, K chunks {lhi*8..+7} and {16+lhi*8..+7}
    auto ldA = [&](const _Float16* A, int f) -> v16h {
        Frag16 r;
        const _Float16* p = A + (wo * 32 + f * 16 + llo) * PITCH + lhi * 8;
        r.u[0] = *(const v4u*)(p);
        r.u[1] = *(const v4u*)(p + 16);
        return r.h;
    };
    // B frag (32x16 f16): lane holds col llo, K chunks {lhi*8..+7} and {16+lhi*8..+7}
    auto ldB = [&](const _Float16* Bt, int mf) -> v16h {
        Frag16 r;
        const _Float16* p = Bt + (wm * 64 + mf * 16 + llo) * PITCH + lhi * 8;
        r.u[0] = *(const v4u*)(p);
        r.u[1] = *(const v4u*)(p + 16);
        return r.h;
    };

    loadStage(0);
    storeStage(0);

    for (int s = 0; s < KTOT / TK; ++s) {
        __syncthreads();                       // stage s stores visible everywhere
        if (s < KTOT / TK - 1) loadStage(TK * (s + 1));
        if (s < KTOT / TK - 2) {               // prefetch X two stages ahead (near scope)
            const int kb2 = TK * (s + 2);
            const float* p2 = ((kb2 < NCH) ? xrb : xib) +
                              (size_t)((kb2 & (NCH - 1)) + 2 * xk) * HW + xm4 * 4;
            __builtin_prefetch(p2, 0, 3);
        }
        const _Float16* A  = &sA[s & 1][0];
        const _Float16* Bt = &sB[s & 1][0];

        v16h af0 = ldA(A, 0);
        v16h af1 = ldA(A, 1);
#pragma unroll
        for (int mf = 0; mf < 4; ++mf) {
            v16h bf = ldB(Bt, mf);
            acc[0][mf].v = __builtin_amdgcn_wmma_f32_16x16x32_f16(
                false, af0, false, bf, (short)0, acc[0][mf].v, false, false);
            acc[1][mf].v = __builtin_amdgcn_wmma_f32_16x16x32_f16(
                false, af1, false, bf, (short)0, acc[1][mf].v, false, false);
        }
        if (s < KTOT / TK - 1) storeStage((s + 1) & 1);
    }

    // Epilogue: D layout -> lane llo gives m, VGPR v gives o = base + v (+8 for lanes 16-31)
    const int shalf = blockIdx.x >> 1;               // 0 = real out, 1 = imag out
#pragma unroll
    for (int f = 0; f < 2; ++f) {
        const int o0 = obase + wo * 32 + f * 16 + lhi * 8;
        const int n0 = o0 & (NCH - 1);
#pragma unroll
        for (int mf = 0; mf < 4; ++mf) {
            const int m = mbase + wm * 64 + mf * 16 + llo;
            const int b = m >> 16;
            const int r = m & (HW - 1);
            const size_t base = ((size_t)shalf * NB + b) * ((size_t)NCH * HW) + r;
#pragma unroll
            for (int v = 0; v < 8; ++v)
                out[base + (size_t)(n0 + v) * HW] = acc[f][mf].f[v];
        }
    }
}

extern "C" void kernel_launch(void* const* d_in, const int* in_sizes, int n_in,
                              void* d_out, int out_size, void* d_ws, size_t ws_size,
                              hipStream_t stream) {
    (void)in_sizes; (void)n_in; (void)out_size; (void)d_ws; (void)ws_size;
    const float* xr = (const float*)d_in[0];   // adc_real  (B,N,H,W) f32
    const float* xi = (const float*)d_in[1];   // adc_imag  (B,N,H,W) f32
    const float* w  = (const float*)d_in[2];   // weight    (512,512) f32
    float* out = (float*)d_out;                // (2,B,N,H,W) f32

    dim3 grid(KTOT / TM /*4 o-tiles, fastest -> share X slab in L2*/,
              (NB * HW) / TN /*2048 m-tiles*/, 1);
    fft_dft_gemm_f16wmma<<<grid, 256, 0, stream>>>(xr, xi, w, out);
}
